// MixtureOfExperts_52974126629144
// MI455X (gfx1250) — matrix operand order repaired and checked by reading, not compiled, not run on hardware
//
#include <hip/hip_runtime.h>
#include <hip/hip_bf16.h>

// ---------------- Problem constants (from reference) ----------------
#define T_TOK 4096   // B*S
#define DIM_D 1024
#define DIM_H 2048
#define DIM_O 1024
#define N_EXP 8

#define MTILE 32                 // tokens per FFN block (2 WMMA row tiles)
#define SMEM_BYTES (MTILE*DIM_D*2 /*X*/ + MTILE*256*2 /*act*/ + MTILE*4 /*tok*/ + MTILE*4 /*gate*/)

// ---------------- Types for WMMA ----------------
typedef __attribute__((ext_vector_type(16))) __bf16         bf16x16;
typedef __attribute__((ext_vector_type(8)))  float          f32x8;
typedef __attribute__((ext_vector_type(4)))  float          f32x4;
typedef __attribute__((ext_vector_type(8)))  unsigned short u16x8;

union Frag16 {
    u16x8   h[2];   // two 16-byte halves
    bf16x16 v;      // WMMA operand view
};

__device__ __forceinline__ unsigned short f2bf(float f) {
    // round-to-nearest-even f32 -> bf16
    unsigned u = __float_as_uint(f);
    unsigned r = (u + 0x7FFFu + ((u >> 16) & 1u)) >> 16;
    return (unsigned short)r;
}

// ---------------- Kernel 0: zero output + expert counters ----------------
__global__ void moe_zero_kernel(float* __restrict__ out, int n, int* __restrict__ counts) {
    int i = blockIdx.x * blockDim.x + threadIdx.x;
    if (i < n) out[i] = 0.0f;
    if (i < N_EXP) counts[i] = 0;
}

// ---------------- Kernel 1: fp32 (E,d1,d2) -> bf16 transposed (E,d2,d1) ----------------
__global__ void moe_cvt_t_kernel(const float* __restrict__ in, unsigned short* __restrict__ out,
                                 int d1, int d2) {
    size_t plane  = (size_t)d1 * d2;
    size_t total  = (size_t)N_EXP * plane;
    size_t stride = (size_t)gridDim.x * blockDim.x;
    for (size_t i = (size_t)blockIdx.x * blockDim.x + threadIdx.x; i < total; i += stride) {
        int e  = (int)(i / plane);
        int r  = (int)(i - (size_t)e * plane);
        int j2 = r / d1;             // output row  (size d2)
        int j1 = r - j2 * d1;        // output col  (size d1)
        out[i] = f2bf(in[(size_t)e * plane + (size_t)j1 * d2 + j2]);
    }
}

// ---------------- Kernel 2: router (softmax top-2, build expert lists) ----------------
__global__ __launch_bounds__(256) void moe_router_kernel(
        const float* __restrict__ x, const float* __restrict__ rw, const float* __restrict__ rb,
        int* __restrict__ counts, int* __restrict__ tlist, float* __restrict__ gates) {
    int t    = blockIdx.x;
    int wave = threadIdx.x >> 5;
    int lane = threadIdx.x & 31;

    const float* xr = x  + (size_t)t    * DIM_D;
    const float* wr = rw + (size_t)wave * DIM_D;

    float p = 0.0f;
    for (int d = lane; d < DIM_D; d += 32) p += xr[d] * wr[d];
#pragma unroll
    for (int off = 16; off > 0; off >>= 1) p += __shfl_xor(p, off, 32);

    __shared__ float lg[N_EXP];
    if (lane == 0) lg[wave] = p + rb[wave];
    __syncthreads();

    if (threadIdx.x == 0) {
        float l0 = lg[0]; int i0 = 0;
        for (int e = 1; e < N_EXP; ++e) if (lg[e] > l0) { l0 = lg[e]; i0 = e; }
        float l1 = -3.4e38f; int i1 = 0;
        for (int e = 0; e < N_EXP; ++e) if (e != i0 && lg[e] > l1) { l1 = lg[e]; i1 = e; }
        // renormalized top-2 softmax: denominator over all 8 cancels
        float p1  = __expf(l1 - l0);
        float inv = 1.0f / (1.0f + p1);
        float w0  = inv, w1 = p1 * inv;

        int s0 = atomicAdd(counts + i0, 1);
        tlist[i0 * T_TOK + s0] = t;  gates[i0 * T_TOK + s0] = w0;
        int s1 = atomicAdd(counts + i1, 1);
        tlist[i1 * T_TOK + s1] = t;  gates[i1 * T_TOK + s1] = w1;
    }
}

// ---------------- Kernel 3: fused grouped-GEMM expert FFN ----------------
// Per block: one expert, one 32-token gathered tile (2 WMMA row tiles), 512 threads / 16 waves.
//   X(32xD) staged bf16 in LDS (dynamic, 64KB). For each 256-wide H chunk:
//     stage A: wave w owns H-column tile w: per k-step load B1/B3 once, both A row halves,
//              issue 4 WMMAs (h,g x 2 M-halves); silu(h)*g -> bf16 act chunk (32x256) in LDS.
//     stage B: wave w owns O cols [64w,64w+64) x both M-halves: per k-step 2 A + 4 B fragment
//              loads feed 8 WMMAs; B(W2) reused across M-halves.
//   Epilogue: gate-scale rows, atomicAdd (exactly 2 commutative f32 adds per out element).
__global__ __launch_bounds__(512) void moe_ffn_kernel(
        const float* __restrict__ x,
        const unsigned short* __restrict__ W1t,   // (E, H, D) bf16
        const unsigned short* __restrict__ W3t,   // (E, H, D) bf16
        const unsigned short* __restrict__ W2t,   // (E, O, H) bf16
        const int* __restrict__ counts, const int* __restrict__ tlist,
        const float* __restrict__ gates, float* __restrict__ out) {
    int e    = blockIdx.y;
    int tile = blockIdx.x;
    int cnt  = counts[e];
    if (tile * MTILE >= cnt) return;

    extern __shared__ __align__(16) char smem[];
    unsigned short* lds_x   = (unsigned short*)smem;                           // 32x1024 bf16
    unsigned short* lds_act = (unsigned short*)(smem + MTILE * DIM_D * 2);     // 32x256  bf16
    int*            lds_tok = (int*)  (smem + MTILE * DIM_D * 2 + MTILE * 256 * 2);
    float*          lds_gate= (float*)(smem + MTILE * DIM_D * 2 + MTILE * 256 * 2 + MTILE * 4);

    int tid = threadIdx.x;
    if (tid < MTILE) {
        int idx = tile * MTILE + tid;
        int tok = (idx < cnt) ? tlist[e * T_TOK + idx] : -1;
        lds_tok[tid]  = tok;
        lds_gate[tid] = (idx < cnt) ? gates[e * T_TOK + idx] : 0.0f;
    }
    __syncthreads();

    // gather X tile (fp32 -> bf16), vectorized: float4 load -> 4x bf16 packed 8B LDS store
    for (int i = tid; i < MTILE * (DIM_D / 4); i += 512) {
        int m  = i >> 8;            // DIM_D/4 = 256 float4 per row
        int c4 = i & 255;
        int tok = lds_tok[m];
        f32x4 v = {0.f, 0.f, 0.f, 0.f};
        if (tok >= 0) v = ((const f32x4*)(x + (size_t)tok * DIM_D))[c4];
        union { unsigned short s[4]; unsigned long long u; } pk;
        pk.s[0] = f2bf(v.x); pk.s[1] = f2bf(v.y); pk.s[2] = f2bf(v.z); pk.s[3] = f2bf(v.w);
        ((unsigned long long*)lds_x)[(size_t)m * (DIM_D / 4) + c4] = pk.u;
    }
    __syncthreads();

    int wave  = tid >> 5;           // 0..15
    int lane  = tid & 31;
    int lhalf = lane >> 4;          // 0 | 1
    int l16   = lane & 15;

    f32x8 oacc[4][2];               // [n-tile][m-half]
#pragma unroll
    for (int i = 0; i < 4; ++i)
#pragma unroll
        for (int hh = 0; hh < 2; ++hh)
#pragma unroll
            for (int j = 0; j < 8; ++j) oacc[i][hh][j] = 0.0f;

    for (int hc = 0; hc < DIM_H / 256; ++hc) {
        // ---------------- stage A ----------------
        int ncol = hc * 256 + wave * 16 + l16;     // global H column (B col for WMMA #1)
        f32x8 acch0 = {0.f,0.f,0.f,0.f,0.f,0.f,0.f,0.f};
        f32x8 accg0 = {0.f,0.f,0.f,0.f,0.f,0.f,0.f,0.f};
        f32x8 acch1 = {0.f,0.f,0.f,0.f,0.f,0.f,0.f,0.f};
        f32x8 accg1 = {0.f,0.f,0.f,0.f,0.f,0.f,0.f,0.f};
        const unsigned short* w1r = W1t + ((size_t)e * DIM_H + ncol) * DIM_D + 8 * lhalf;
        const unsigned short* w3r = W3t + ((size_t)e * DIM_H + ncol) * DIM_D + 8 * lhalf;

        for (int k0 = 0; k0 < DIM_D; k0 += 32) {
            __builtin_prefetch(w1r + k0 + 256, 0, 1);   // global_prefetch_b8
            __builtin_prefetch(w3r + k0 + 256, 0, 1);
            Frag16 a0, a1, b1, b3;
            // A 16-bit 16x32 layout: lane holds K in {base..+7, base+16..+23}, base=8*lhalf
            const u16x8* ap0 = (const u16x8*)&lds_x[l16 * DIM_D + k0 + 8 * lhalf];
            const u16x8* ap1 = (const u16x8*)&lds_x[(16 + l16) * DIM_D + k0 + 8 * lhalf];
            a0.h[0] = ap0[0];  a0.h[1] = ap0[2];
            a1.h[0] = ap1[0];  a1.h[1] = ap1[2];
            // B 32x16 layout: lane = column, K in {base..+7, base+16..+23}, base=8*lhalf
            const u16x8* p1 = (const u16x8*)(w1r + k0);
            const u16x8* p3 = (const u16x8*)(w3r + k0);
            b1.h[0] = p1[0]; b1.h[1] = p1[2];
            b3.h[0] = p3[0]; b3.h[1] = p3[2];
            acch0 = __builtin_amdgcn_wmma_f32_16x16x32_bf16(false, a0.v, false, b1.v,
                                                            (short)0, acch0, false, false);
            acch1 = __builtin_amdgcn_wmma_f32_16x16x32_bf16(false, a1.v, false, b1.v,
                                                            (short)0, acch1, false, false);
            accg0 = __builtin_amdgcn_wmma_f32_16x16x32_bf16(false, a0.v, false, b3.v,
                                                            (short)0, accg0, false, false);
            accg1 = __builtin_amdgcn_wmma_f32_16x16x32_bf16(false, a1.v, false, b3.v,
                                                            (short)0, accg1, false, false);
        }

        __syncthreads();   // previous stage-B readers of lds_act are done
#pragma unroll
        for (int v = 0; v < 8; ++v) {
            int m0 = v + 8 * lhalf;                // C/D layout: VGPR v, lane half -> row M
            float h0 = acch0[v];
            float s0 = h0 / (1.0f + __expf(-h0));
            lds_act[m0 * 256 + wave * 16 + l16] = f2bf(s0 * accg0[v]);
            float h1 = acch1[v];
            float s1 = h1 / (1.0f + __expf(-h1));
            lds_act[(16 + m0) * 256 + wave * 16 + l16] = f2bf(s1 * accg1[v]);
        }
        __syncthreads();

        // ---------------- stage B ----------------
        const unsigned short* w2base = W2t + ((size_t)e * DIM_O + wave * 64 + l16) * DIM_H
                                       + hc * 256 + 8 * lhalf;
        for (int ks = 0; ks < 256; ks += 32) {
            Frag16 a0, a1;
            const u16x8* ap0 = (const u16x8*)&lds_act[l16 * 256 + ks + 8 * lhalf];
            const u16x8* ap1 = (const u16x8*)&lds_act[(16 + l16) * 256 + ks + 8 * lhalf];
            a0.h[0] = ap0[0];  a0.h[1] = ap0[2];
            a1.h[0] = ap1[0];  a1.h[1] = ap1[2];
#pragma unroll
            for (int nt = 0; nt < 4; ++nt) {
                Frag16 b;
                const u16x8* bp = (const u16x8*)(w2base + (size_t)nt * 16 * DIM_H + ks);
                b.h[0] = bp[0];
                b.h[1] = bp[2];
                oacc[nt][0] = __builtin_amdgcn_wmma_f32_16x16x32_bf16(false, a0.v, false, b.v,
                                                                      (short)0, oacc[nt][0],
                                                                      false, false);
                oacc[nt][1] = __builtin_amdgcn_wmma_f32_16x16x32_bf16(false, a1.v, false, b.v,
                                                                      (short)0, oacc[nt][1],
                                                                      false, false);
            }
        }
    }

    // ---------------- epilogue: gate-scale + scatter-accumulate ----------------
#pragma unroll
    for (int nt = 0; nt < 4; ++nt) {
        int ocol = wave * 64 + nt * 16 + l16;
#pragma unroll
        for (int hh = 0; hh < 2; ++hh) {
#pragma unroll
            for (int v = 0; v < 8; ++v) {
                int m   = hh * 16 + v + 8 * lhalf;
                int tok = lds_tok[m];
                if (tok >= 0)
                    atomicAdd(&out[(size_t)tok * DIM_O + ocol], oacc[nt][hh][v] * lds_gate[m]);
            }
        }
    }
}

// ---------------- Launch ----------------
extern "C" void kernel_launch(void* const* d_in, const int* in_sizes, int n_in,
                              void* d_out, int out_size, void* d_ws, size_t ws_size,
                              hipStream_t stream) {
    const float* x  = (const float*)d_in[0];
    const float* rw = (const float*)d_in[1];
    const float* rb = (const float*)d_in[2];
    const float* W1 = (const float*)d_in[3];
    const float* W3 = (const float*)d_in[4];
    const float* W2 = (const float*)d_in[5];
    float* out = (float*)d_out;

    // workspace layout
    char*  ws     = (char*)d_ws;
    int*   counts = (int*)ws;                              // 32 B (256 B aligned region)
    int*   tlist  = (int*)(ws + 256);                      // E*T ints  = 128 KB
    float* gates  = (float*)(ws + 256 + (size_t)N_EXP * T_TOK * 4);
    size_t woff   = 256 + (size_t)N_EXP * T_TOK * 4 * 2;
    woff = (woff + 255) & ~(size_t)255;
    unsigned short* W1t = (unsigned short*)(ws + woff); woff += (size_t)N_EXP * DIM_H * DIM_D * 2;
    unsigned short* W3t = (unsigned short*)(ws + woff); woff += (size_t)N_EXP * DIM_H * DIM_D * 2;
    unsigned short* W2t = (unsigned short*)(ws + woff); woff += (size_t)N_EXP * DIM_O * DIM_H * 2;

    int nOut = T_TOK * DIM_O;
    moe_zero_kernel<<<(nOut + 255) / 256, 256, 0, stream>>>(out, nOut, counts);

    moe_cvt_t_kernel<<<8192, 256, 0, stream>>>(W1, W1t, DIM_D, DIM_H);
    moe_cvt_t_kernel<<<8192, 256, 0, stream>>>(W3, W3t, DIM_D, DIM_H);
    moe_cvt_t_kernel<<<8192, 256, 0, stream>>>(W2, W2t, DIM_H, DIM_O);

    moe_router_kernel<<<T_TOK, 256, 0, stream>>>(x, rw, rb, counts, tlist, gates);

    dim3 grid(T_TOK / MTILE, N_EXP);
    moe_ffn_kernel<<<grid, 512, SMEM_BYTES, stream>>>(x, W1t, W3t, W2t,
                                                      counts, tlist, gates, out);
}
